// RouterCompoundFast_1984274891214
// MI455X (gfx1250) — matrix operand order repaired and checked by reading, not compiled, not run on hardware
//
#include <hip/hip_runtime.h>
#include <hip/hip_bf16.h>
#include <math.h>

// ---------------- problem constants (match reference) ----------------
#define HIDDEN   1024
#define N_ROUTED 16
#define INNER    4
#define BIGGER   64
#define ROWS     (BIGGER * INNER)   // 256 rows per expert matrix
#define ACTI     4
#define TOTAL    6                  // sum(PATTERN)
#define BS       128
#define MAXP     128                // max (token,slot) pairs per expert (<= BS)
#define KSTEPS   (HIDDEN / 32)      // 32 WMMA K-steps of 32
#define A_STRIDE 1032               // halves; padded row stride -> conflict-free b128 LDS reads

typedef __attribute__((ext_vector_type(16))) _Float16 v16h;
typedef __attribute__((ext_vector_type(8)))  _Float16 v8h;
typedef __attribute__((ext_vector_type(8)))  float    v8f;
typedef __attribute__((ext_vector_type(4)))  float    v4f;

// ---------------------------------------------------------------------
// Kernel 0: reset per-expert pair counters (workspace is not re-poisoned)
// ---------------------------------------------------------------------
__global__ void rcf_zero_counts(int* __restrict__ counts) {
    if (threadIdx.x < N_ROUTED) counts[threadIdx.x] = 0;
}

// ---------------------------------------------------------------------
// Kernel 1: routing. One block per token (128 threads = 4 waves).
//   logits = x @ Wg^T, softmax, top-4 (strict '>' => jax tie-break),
//   normalized weights; bucket (token,slot) pairs per expert.
// ---------------------------------------------------------------------
__global__ __launch_bounds__(128, 4)
void rcf_routing(const float* __restrict__ x,
                 const float* __restrict__ gw,
                 int*   __restrict__ counts,
                 int*   __restrict__ pairs,
                 float* __restrict__ flatW)
{
    __shared__ float partial[128];
    __shared__ float logits[N_ROUTED];

    const int token = blockIdx.x;
    const int t     = threadIdx.x;
    const int e     = t >> 3;       // 16 experts x 8 threads
    const int sub   = t & 7;

    const float* xs = x  + (size_t)token * HIDDEN + sub * (HIDDEN / 8);
    const float* ws = gw + (size_t)e     * HIDDEN + sub * (HIDDEN / 8);
    float s = 0.f;
    #pragma unroll 4
    for (int j = 0; j < HIDDEN / 8; j += 4) {
        v4f a = *(const v4f*)(xs + j);
        v4f b = *(const v4f*)(ws + j);
        s += a[0]*b[0] + a[1]*b[1] + a[2]*b[2] + a[3]*b[3];
    }
    partial[t] = s;
    __syncthreads();

    if (t < N_ROUTED) {                  // deterministic fixed-order tree
        float acc = 0.f;
        #pragma unroll
        for (int k = 0; k < 8; ++k) acc += partial[t * 8 + k];
        logits[t] = acc;
    }
    __syncthreads();

    if (t == 0) {
        float mx = logits[0];
        #pragma unroll
        for (int i = 1; i < N_ROUTED; ++i) mx = fmaxf(mx, logits[i]);
        float p[N_ROUTED];
        float denom = 0.f;
        #pragma unroll
        for (int i = 0; i < N_ROUTED; ++i) { p[i] = __expf(logits[i] - mx); denom += p[i]; }
        const float inv = 1.f / denom;
        #pragma unroll
        for (int i = 0; i < N_ROUTED; ++i) p[i] *= inv;

        int   sel[ACTI]; float wsel[ACTI];
        bool  used[N_ROUTED];
        #pragma unroll
        for (int i = 0; i < N_ROUTED; ++i) used[i] = false;
        float wsum = 0.f;
        #pragma unroll
        for (int slot = 0; slot < ACTI; ++slot) {
            float best = -INFINITY; int bi = 0;
            for (int i = 0; i < N_ROUTED; ++i)
                if (!used[i] && p[i] > best) { best = p[i]; bi = i; }
            used[bi] = true; sel[slot] = bi; wsel[slot] = best; wsum += best;
        }
        const float invw = 1.f / wsum;
        #pragma unroll
        for (int slot = 0; slot < ACTI; ++slot) {
            const int   ex = sel[slot];
            const float w  = wsel[slot] * invw;
            const int pos  = atomicAdd(&counts[ex], 1);   // order-independent downstream
            pairs[ex * MAXP + pos]        = token * ACTI + slot;
            flatW[token * ACTI + slot]    = w;
        }
    }
}

// ---------------------------------------------------------------------
// Kernel 2: grouped expert GEMM via WMMA + score + inner top-2 + scatter.
// grid = (16 experts, 8 token-tiles), block = 256 threads (8 wave32).
// Each wave owns two 16-row N-tiles; one A fragment feeds 4 WMMAs/k-step.
// ---------------------------------------------------------------------
__global__ __launch_bounds__(256, 1)
void rcf_moe_wmma(const float* __restrict__ x,
                  const float* __restrict__ gateW,
                  const float* __restrict__ upW,
                  const int*   __restrict__ counts,
                  const int*   __restrict__ pairs,
                  const float* __restrict__ flatW,
                  float* __restrict__ out)
{
    const int e    = blockIdx.x;
    const int tile = blockIdx.y;
    const int cnt  = counts[e];
    const int rem  = cnt - tile * 16;
    if (rem <= 0) return;                      // uniform early exit (before barriers)
    const int np = rem > 16 ? 16 : rem;

    __shared__ _Float16 Atile[16 * A_STRIDE];  // 16 tokens x 1024 f16 (padded)
    __shared__ float    sc[16][ROWS];          // scores [pair][row]
    __shared__ float    means[16][INNER];
    __shared__ int      sTok[16], sSlot[16];
    __shared__ float    sW[16];

    const int tid = threadIdx.x;

    // ---- pair metadata ----
    if (tid < 16) {
        if (tid < np) {
            const int pid = pairs[e * MAXP + tile * 16 + tid];
            sTok[tid]  = pid >> 2;
            sSlot[tid] = pid & 3;
            sW[tid]    = flatW[pid];
        } else {
            sTok[tid] = -1; sSlot[tid] = 0; sW[tid] = 0.f;
        }
    }
    __syncthreads();

    // ---- stage A: x rows f32 -> f16 into LDS (16 threads per pair, 64 floats each) ----
    {
        const int pi  = tid >> 4;
        const int col = (tid & 15) * 64;
        _Float16* dst = &Atile[pi * A_STRIDE + col];
        if (sTok[pi] >= 0) {
            const float* src = x + (size_t)sTok[pi] * HIDDEN + col;
            #pragma unroll
            for (int c = 0; c < 8; ++c) {
                v4f f0 = *(const v4f*)(src + c * 8);
                v4f f1 = *(const v4f*)(src + c * 8 + 4);
                v8h h;
                #pragma unroll
                for (int j = 0; j < 4; ++j) { h[j] = (_Float16)f0[j]; h[j + 4] = (_Float16)f1[j]; }
                *(v8h*)(dst + c * 8) = h;          // ds_store_b128
            }
        } else {
            v8h z;
            #pragma unroll
            for (int j = 0; j < 8; ++j) z[j] = (_Float16)0.f;
            #pragma unroll
            for (int c = 0; c < 8; ++c) *(v8h*)(dst + c * 8) = z;
        }
    }
    __syncthreads();

    // ---- main WMMA loop ----
    const int wid    = tid >> 5;
    const int lane   = tid & 31;
    const int nrow   = lane & 15;    // N index within tile / A-row M index
    const int hiLane = lane >> 4;    // 0 or 1

    const _Float16* aRow = &Atile[nrow * A_STRIDE];

    const int rowBase0 = (wid * 2 + 0) * 16;
    const int rowBase1 = (wid * 2 + 1) * 16;
    const float* g0 = gateW + ((size_t)e * ROWS + rowBase0 + nrow) * HIDDEN;
    const float* u0 = upW   + ((size_t)e * ROWS + rowBase0 + nrow) * HIDDEN;
    const float* g1 = gateW + ((size_t)e * ROWS + rowBase1 + nrow) * HIDDEN;
    const float* u1 = upW   + ((size_t)e * ROWS + rowBase1 + nrow) * HIDDEN;

    v8f gacc0, uacc0, gacc1, uacc1;
    #pragma unroll
    for (int i = 0; i < 8; ++i) { gacc0[i] = 0.f; uacc0[i] = 0.f; gacc1[i] = 0.f; uacc1[i] = 0.f; }

    for (int ks = 0; ks < KSTEPS; ++ks) {
        const int kb = ks * 32;
        // A fragment: lanes 0-15 K={0..15}, lanes 16-31 K={16..31}
        v8h alo = *(const v8h*)(aRow + kb + hiLane * 16);        // ds_load_b128
        v8h ahi = *(const v8h*)(aRow + kb + hiLane * 16 + 8);   // ds_load_b128
        v16h a;
        #pragma unroll
        for (int i = 0; i < 8; ++i) { a[i] = alo[i]; a[i + 8] = ahi[i]; }

        // B fragments: lane = N column, 16 contiguous K per lane (fp32 -> f16 cvt)
        const int kcol = kb + hiLane * 16;
        v16h bg0, bu0, bg1, bu1;
        #pragma unroll
        for (int c = 0; c < 4; ++c) {
            v4f fg0 = *(const v4f*)(g0 + kcol + c * 4);         // global_load_b128 x4
            v4f fu0 = *(const v4f*)(u0 + kcol + c * 4);
            v4f fg1 = *(const v4f*)(g1 + kcol + c * 4);
            v4f fu1 = *(const v4f*)(u1 + kcol + c * 4);
            #pragma unroll
            for (int j = 0; j < 4; ++j) {
                bg0[c * 4 + j] = (_Float16)fg0[j];
                bu0[c * 4 + j] = (_Float16)fu0[j];
                bg1[c * 4 + j] = (_Float16)fg1[j];
                bu1[c * 4 + j] = (_Float16)fu1[j];
            }
        }
        __builtin_prefetch(g0 + kcol + 32, 0, 1);               // global_prefetch_b8
        __builtin_prefetch(u0 + kcol + 32, 0, 1);
        __builtin_prefetch(g1 + kcol + 32, 0, 1);
        __builtin_prefetch(u1 + kcol + 32, 0, 1);

        gacc0 = __builtin_amdgcn_wmma_f32_16x16x32_f16(false, a, false, bg0,
                                                       (short)0, gacc0, false, false);
        uacc0 = __builtin_amdgcn_wmma_f32_16x16x32_f16(false, a, false, bu0,
                                                       (short)0, uacc0, false, false);
        gacc1 = __builtin_amdgcn_wmma_f32_16x16x32_f16(false, a, false, bg1,
                                                       (short)0, gacc1, false, false);
        uacc1 = __builtin_amdgcn_wmma_f32_16x16x32_f16(false, a, false, bu1,
                                                       (short)0, uacc1, false, false);
    }

    // scores: |up * silu(gate)| with fast rcp; C layout: VGPR r -> M = r + 8*hiLane, N = nrow
    #pragma unroll
    for (int r = 0; r < 8; ++r) {
        const int m = r + hiLane * 8;
        {
            const float g = gacc0[r], u = uacc0[r];
            const float sig = __builtin_amdgcn_rcpf(1.f + __expf(-g));   // v_rcp_f32
            sc[m][rowBase0 + nrow] = fabsf(u * g * sig);
        }
        {
            const float g = gacc1[r], u = uacc1[r];
            const float sig = __builtin_amdgcn_rcpf(1.f + __expf(-g));
            sc[m][rowBase1 + nrow] = fabsf(u * g * sig);
        }
    }
    __syncthreads();

    // ---- mean over 64 rows per inner group (fixed order => deterministic) ----
    if (tid < 16 * INNER) {
        const int m = tid >> 2, g = tid & 3;
        float s = 0.f;
        for (int r = 0; r < BIGGER; ++r) s += sc[m][g * BIGGER + r];
        means[m][g] = s * (1.f / BIGGER);
    }
    __syncthreads();

    // ---- inner top-2 + static masked-select scatter ----
    if (tid < np) {
        float best0 = -INFINITY; int i0 = 0;
        #pragma unroll
        for (int g = 0; g < INNER; ++g) { const float v = means[tid][g]; if (v > best0) { best0 = v; i0 = g; } }
        float best1 = -INFINITY; int i1 = 0;
        #pragma unroll
        for (int g = 0; g < INNER; ++g) {
            if (g == i0) continue;
            const float v = means[tid][g]; if (v > best1) { best1 = v; i1 = g; }
        }
        const int   token = sTok[tid];
        const int   slot  = sSlot[tid];
        const float w     = sW[tid];
        const float id0   = (float)(e * INNER + i0);
        const float id1   = (float)(e * INNER + i1);
        float* outW = out;                 // [128,6] weights
        float* outI = out + BS * TOTAL;    // [128,6] ids (emitted as float values)
        const int base = token * TOTAL;
        if      (slot == 0) { outW[base + 0] = w; outW[base + 1] = w; outI[base + 0] = id0; outI[base + 1] = id1; }
        else if (slot == 1) { outW[base + 2] = w; outW[base + 3] = w; outI[base + 2] = id0; outI[base + 3] = id1; }
        else if (slot == 2) { outW[base + 4] = w; outI[base + 4] = id0; }
        else                { outW[base + 5] = w; outI[base + 5] = id0; }
    }
}

// ---------------------------------------------------------------------
extern "C" void kernel_launch(void* const* d_in, const int* in_sizes, int n_in,
                              void* d_out, int out_size, void* d_ws, size_t ws_size,
                              hipStream_t stream) {
    (void)in_sizes; (void)n_in; (void)out_size; (void)ws_size;
    const float* x     = (const float*)d_in[0];   // [128, 1024]
    const float* gw    = (const float*)d_in[1];   // [16, 1024]
    const float* gateW = (const float*)d_in[2];   // [16, 256, 1024]
    const float* upW   = (const float*)d_in[3];   // [16, 256, 1024]
    float* out = (float*)d_out;                   // [128,6] weights ++ [128,6] ids

    int*   counts = (int*)d_ws;                   // 16 ints
    int*   pairsB = counts + N_ROUTED;            // 16*128 ints
    float* flatW  = (float*)(pairsB + N_ROUTED * MAXP);  // 512 floats

    rcf_zero_counts<<<1, 32, 0, stream>>>(counts);
    rcf_routing<<<BS, 128, 0, stream>>>(x, gw, counts, pairsB, flatW);
    rcf_moe_wmma<<<dim3(N_ROUTED, MAXP / 16), 256, 0, stream>>>(
        x, gateW, upW, counts, pairsB, flatW, out);
}
